// SAGPool_29351806501361
// MI455X (gfx1250) — matrix operand order repaired and checked by reading, not compiled, not run on hardware
//
#include <hip/hip_runtime.h>
#include <hip/hip_bf16.h>
#include <math.h>

typedef __attribute__((ext_vector_type(16))) _Float16 v16h;
typedef __attribute__((ext_vector_type(8)))  float    v8f;

#define B_GRAPHS 256
#define NPG      512
#define NTOT     (B_GRAPHS * NPG)     // 131072
#define DIM      256
#define EPG      16384                // edges per graph
#define ETOT     (EPG * B_GRAPHS)     // 4194304
#define KSEL     410                  // ceil(0.8 * 512)

// ---------------------------------------------------------------------------
// Kernel 1: h = x @ W  via V_WMMA_F32_16X16X32_F16.
// One wave = 16 rows of x. A-tile: 16x32 (f32 -> f16). B-tile: 32x16 with only
// column N=0 holding W[kchunk]. Accumulate over 8 K-chunks (D=256).
// C layout: column N=0 lives in lane 0 (M=0..7) and lane 16 (M=8..15).
// ---------------------------------------------------------------------------
__global__ __launch_bounds__(256)
void gcn_matvec_wmma(const float* __restrict__ x,
                     const float* __restrict__ W,
                     float* __restrict__ h) {
    const int lane    = threadIdx.x & 31;
    const int wave    = threadIdx.x >> 5;
    const int r0      = (blockIdx.x * 8 + wave) * 16;   // first row of tile
    const int rowLane = lane & 15;
    const int row     = r0 + rowLane;
    const bool hiHalf = (lane >= 16);

    v8f c = {};
    #pragma unroll
    for (int kc = 0; kc < DIM; kc += 32) {
        // ---- A fragment (16x32 f16), per-lane layout from ISA 7.12.2 ----
        // lanes 0-15 : elems 0..7 = K kc+0..7,  elems 8..15 = K kc+16..23
        // lanes 16-31: elems 0..7 = K kc+8..15, elems 8..15 = K kc+24..31
        const float* xr = x + (size_t)row * DIM + kc + (hiHalf ? 8 : 0);
        v16h a;
        #pragma unroll
        for (int e = 0; e < 8; ++e) a[e]     = (_Float16)xr[e];
        #pragma unroll
        for (int e = 0; e < 8; ++e) a[8 + e] = (_Float16)xr[16 + e];

        // ---- B fragment (32x16 f16): only column N=0 is W ----
        // lane 0  : K kc+0..7,16..23 ; lane 16 : K kc+8..15,24..31 ; all other lanes 0
        v16h b = {};
        if (rowLane == 0) {
            const float* wp = W + kc + (hiHalf ? 8 : 0);
            #pragma unroll
            for (int e = 0; e < 8; ++e) b[e] = (_Float16)wp[e];
            #pragma unroll
            for (int e = 0; e < 8; ++e) b[8 + e] = (_Float16)wp[16 + e];
        }

        c = __builtin_amdgcn_wmma_f32_16x16x32_f16(
                /*neg_a=*/false, a, /*neg_b=*/false, b,
                /*c_mod=*/(short)0, c, /*reuse_a=*/false, /*reuse_b=*/false);
    }

    // Column N=0 extraction.
    if (rowLane == 0) {
        const int base = r0 + (hiHalf ? 8 : 0);
        #pragma unroll
        for (int r = 0; r < 8; ++r) h[base + r] = c[r];
    }
}

// ---------------------------------------------------------------------------
// Kernel 2: per-graph degree count, symmetric norm, edge aggregation (all in
// LDS via ds atomics), score = agg + dinv^2*h + b, then rank-based top-K
// selection (matches jax top_k tie-breaking: value desc, index asc).
// One workgroup (256 threads) per graph: 512 nodes, 16384 edges.
// ---------------------------------------------------------------------------
__global__ __launch_bounds__(256)
void gcn_score_select(const int* __restrict__ ei,      // [2, E] flat
                      const float* __restrict__ h,
                      const float* __restrict__ bptr,
                      float* __restrict__ score_out,   // [N]
                      int* __restrict__ sel,           // [N]
                      float* __restrict__ scale) {     // [N] tanh(score)
    __shared__ float sH[NPG];
    __shared__ float sDinv[NPG];
    __shared__ int   sCnt[NPG];
    __shared__ float sScore[NPG];

    const int g        = blockIdx.x;
    const int tid      = threadIdx.x;
    const int nodeBase = g * NPG;
    const int* srcE = ei + (size_t)g * EPG;          // graph's src slice
    const int* dstE = ei + (size_t)ETOT + (size_t)g * EPG; // graph's dst slice

    for (int i = tid; i < NPG; i += 256) {
        sH[i]     = h[nodeBase + i];
        sCnt[i]   = 0;
        sScore[i] = 0.0f;
    }
    __syncthreads();

    // degree of dst (col) via LDS int atomics
    for (int e = tid; e < EPG; e += 256)
        atomicAdd(&sCnt[dstE[e] - nodeBase], 1);
    __syncthreads();

    // dinv = rsqrt(deg + 1 self loop)
    for (int i = tid; i < NPG; i += 256)
        sDinv[i] = rsqrtf((float)(sCnt[i] + 1));
    __syncthreads();

    // agg[dst] += dinv[src]*dinv[dst]*h[src] via LDS float atomics
    for (int e = tid; e < EPG; e += 256) {
        const int sl = srcE[e] - nodeBase;
        const int dl = dstE[e] - nodeBase;
        atomicAdd(&sScore[dl], sDinv[sl] * sDinv[dl] * sH[sl]);
    }
    __syncthreads();

    // add self-loop term and bias
    const float bb = bptr[0];
    for (int i = tid; i < NPG; i += 256)
        sScore[i] = sScore[i] + sDinv[i] * sDinv[i] * sH[i] + bb;
    __syncthreads();

    // rank-by-counting top-K selection (order-free: pooling is max/mean)
    for (int i = tid; i < NPG; i += 256) {
        const float si = sScore[i];
        int rank = 0;
        for (int j = 0; j < NPG; ++j) {
            const float sj = sScore[j];
            rank += (sj > si) || ((sj == si) && (j < i));
        }
        score_out[nodeBase + i] = si;
        scale[nodeBase + i]     = tanhf(si);
        sel[nodeBase + i]       = (rank < KSEL) ? 1 : 0;
    }
}

// ---------------------------------------------------------------------------
// Kernel 3: per-graph max || mean pool over selected nodes of x * tanh(score).
// Thread = feature dim (coalesced 1KB row reads); uniform branch on sel[n]
// skips unselected rows. x likely L2-resident (128MB vs 192MB L2).
// ---------------------------------------------------------------------------
__global__ __launch_bounds__(256)
void sag_pool(const float* __restrict__ x,
              const int* __restrict__ sel,
              const float* __restrict__ scale,
              float* __restrict__ pooled) {           // [B, 2D]
    __shared__ int   sSel[NPG];
    __shared__ float sSc[NPG];

    const int g = blockIdx.x;
    const int d = threadIdx.x;   // 0..255

    for (int i = threadIdx.x; i < NPG; i += 256) {
        sSel[i] = sel[g * NPG + i];
        sSc[i]  = scale[g * NPG + i];
    }
    __syncthreads();

    float mx  = -INFINITY;
    float sum = 0.0f;
    const float* xg = x + (size_t)g * NPG * DIM;
    for (int n = 0; n < NPG; ++n) {
        if (sSel[n]) {                                   // uniform across block
            const float v = xg[(size_t)n * DIM + d] * sSc[n];
            mx  = fmaxf(mx, v);
            sum += v;
        }
    }
    pooled[g * (2 * DIM) + d]       = mx;
    pooled[g * (2 * DIM) + DIM + d] = sum * (1.0f / (float)KSEL);
}

// ---------------------------------------------------------------------------
extern "C" void kernel_launch(void* const* d_in, const int* in_sizes, int n_in,
                              void* d_out, int out_size, void* d_ws, size_t ws_size,
                              hipStream_t stream) {
    const float* x   = (const float*)d_in[0];   // [N, D]
    const int*   ei  = (const int*)  d_in[1];   // [2, E]
    // d_in[2] = batch (unused: graphs are equal-size, contiguous)
    const float* W   = (const float*)d_in[3];   // [D, 1]
    const float* b   = (const float*)d_in[4];   // [1]

    float* pooled = (float*)d_out;                          // B * 2D
    float* score  = (float*)d_out + (size_t)B_GRAPHS * 2 * DIM;

    float* h     = (float*)d_ws;                                   // N f32
    int*   sel   = (int*)  ((char*)d_ws + (size_t)NTOT * 4);       // N i32
    float* scale = (float*)((char*)d_ws + (size_t)NTOT * 8);       // N f32

    gcn_matvec_wmma <<<NTOT / 128, 256, 0, stream>>>(x, W, h);
    gcn_score_select<<<B_GRAPHS,   256, 0, stream>>>(ei, h, b, score, sel, scale);
    sag_pool        <<<B_GRAPHS,   256, 0, stream>>>(x, sel, scale, pooled);
}
